// MQA_74560632259155
// MI455X (gfx1250) — matrix operand order repaired and checked by reading, not compiled, not run on hardware
//
#include <hip/hip_runtime.h>
#include <math.h>

typedef __attribute__((ext_vector_type(16))) _Float16 v16h;
typedef __attribute__((ext_vector_type(8)))  _Float16 v8h;
typedef __attribute__((ext_vector_type(8)))  float    v8f;

static __device__ __forceinline__ v16h cat8(v8h a, v8h b) {
    return __builtin_shufflevector(a, b, 0,1,2,3,4,5,6,7,8,9,10,11,12,13,14,15);
}

// Async global->LDS copy of 16B per lane (ASYNCcnt-tracked, no VGPR data path).
static __device__ __forceinline__ void async_b128(unsigned lds, const void* gp) {
    asm volatile("global_load_async_to_lds_b128 %0, %1, off"
                 :: "v"(lds), "v"(gp) : "memory");
}
static __device__ __forceinline__ unsigned lds_off(const void* p) {
    return (unsigned)(size_t)p;   // low 32 bits of flat LDS aperture = LDS offset
}
static __device__ __forceinline__ void wait_async4() {
    asm volatile("s_wait_asynccnt 0x4" ::: "memory");
}
static __device__ __forceinline__ void wait_async0() {
    asm volatile("s_wait_asynccnt 0x0" ::: "memory");
}

// ---------------------------------------------------------------- prep kernels
__global__ __launch_bounds__(256) void convert_f16(const float* __restrict__ X,
                                                   _Float16* __restrict__ XH, int n) {
    int i = blockIdx.x * 256 + threadIdx.x;
    if (i < n) XH[i] = (_Float16)X[i];
}

// W[K][N] (fp32) -> WT[N][K] (f16): idx runs over N*K, writes coalesced.
__global__ __launch_bounds__(256) void transpose_w(const float* __restrict__ W,
                                                   _Float16* __restrict__ WT,
                                                   int K, int N) {
    int idx = blockIdx.x * 256 + threadIdx.x;
    int n = idx / K, k = idx - n * K;
    WT[idx] = (_Float16)W[(size_t)k * N + n];
}

// ---------------------------------------------------------------- f16 GEMM
// C = A[M][K] * BT[N][K]^T + bias ; output f16, optionally stored transposed.
// Block tile 128x128, BK=32, 256 threads = 8 waves (4 in M x 2 in N),
// each wave a 32x64 region = 2x4 WMMA subtiles. Double-buffered LDS, async
// global->LDS loads pipelined one tile ahead, last iteration peeled so the
// steady-state loop is branch-free.
#define LPAD 40   // LDS row stride in halves (80B, 16B aligned, conflict-free)

__global__ __launch_bounds__(256) void gemm_qkv(
    const _Float16* __restrict__ A,   // [M][K]
    const _Float16* __restrict__ BT,  // [N][K]
    const float*    __restrict__ bias,// [N]
    _Float16*       __restrict__ C,   // [M][N] or (transposed) [N][M]
    int M, int N, int K, int transposed)
{
    __shared__ _Float16 Al[2][128 * LPAD];   // 2 x 10 KB
    __shared__ _Float16 Bl[2][128 * LPAD];   // 2 x 10 KB

    const int tid  = threadIdx.x;
    const int lane = tid & 31;
    const int wave = tid >> 5;
    const int lr   = lane & 15;
    const int hi   = lane >> 4;           // 0: lanes 0-15, 1: lanes 16-31
    const int wm   = (wave >> 1) * 32;    // wave M offset in tile
    const int wn   = (wave & 1) * 64;     // wave N offset in tile
    const int gm0  = blockIdx.y * 128;
    const int gn0  = blockIdx.x * 128;

    v8f acc[2][4];
#pragma unroll
    for (int mi = 0; mi < 2; mi++)
#pragma unroll
        for (int ni = 0; ni < 4; ni++)
            acc[mi][ni] = (v8f){0.f,0.f,0.f,0.f,0.f,0.f,0.f,0.f};

    const int srow  = tid >> 1;           // staging: 2 threads per 64B row
    const int spart = (tid & 1) * 16;     // halves

    const _Float16* gA = A  + (size_t)(gm0 + srow) * K + spart;
    const _Float16* gB = BT + (size_t)(gn0 + srow) * K + spart;

    auto stage = [&](int buf, int k0) {   // 4 async 16B loads per lane
        unsigned la = lds_off(&Al[buf][srow * LPAD + spart]);
        unsigned lb = lds_off(&Bl[buf][srow * LPAD + spart]);
        async_b128(la,      gA + k0);
        async_b128(la + 16, gA + k0 + 8);
        async_b128(lb,      gB + k0);
        async_b128(lb + 16, gB + k0 + 8);
    };

    const int aoff = hi ? 8 : 0;          // A frag: K {0-7,16-23} / {8-15,24-31}
    const int boff = hi ? 8 : 0;          // B frag: K {0-7,16-23} / {8-15,24-31}

    auto compute = [&](int buf) {
        v16h af[2], bf[4];
#pragma unroll
        for (int mi = 0; mi < 2; mi++) {
            const _Float16* p = &Al[buf][(wm + mi * 16 + lr) * LPAD + aoff];
            af[mi] = cat8(*(const v8h*)p, *(const v8h*)(p + 16));
        }
#pragma unroll
        for (int ni = 0; ni < 4; ni++) {
            const _Float16* p = &Bl[buf][(wn + ni * 16 + lr) * LPAD + boff];
            bf[ni] = cat8(*(const v8h*)p, *(const v8h*)(p + 16));
        }
#pragma unroll
        for (int mi = 0; mi < 2; mi++)
#pragma unroll
            for (int ni = 0; ni < 4; ni++)
                acc[mi][ni] = __builtin_amdgcn_wmma_f32_16x16x32_f16(
                    false, af[mi], false, bf[ni], (short)0, acc[mi][ni], false, false);
    };

    stage(0, 0);
    int buf = 0;
    for (int k0 = 0; k0 < K - 32; k0 += 32, buf ^= 1) {  // steady state: branch-free
        stage(buf ^ 1, k0 + 32);
        wait_async4();                    // oldest 4 (current buffer) done
        __syncthreads();
        compute(buf);
        __syncthreads();                  // reads done before buf is rewritten
    }
    wait_async0();                        // peeled last tile
    __syncthreads();
    compute(buf);

#pragma unroll
    for (int mi = 0; mi < 2; mi++)
#pragma unroll
        for (int ni = 0; ni < 4; ni++) {
            const int col   = gn0 + wn + ni * 16 + lr;
            const int rbase = gm0 + wm + mi * 16 + hi * 8;
            const float bs  = bias[col];
            if (!transposed) {
#pragma unroll
                for (int r = 0; r < 8; r++)
                    C[(size_t)(rbase + r) * N + col] = (_Float16)(acc[mi][ni][r] + bs);
            } else {
                v8h pk;
#pragma unroll
                for (int r = 0; r < 8; r++)
                    pk[r] = (_Float16)(acc[mi][ni][r] + bs);
                *(v8h*)(C + (size_t)col * M + rbase) = pk;  // 16B-aligned b128 store
            }
        }
}

// ---------------------------------------------------------------- flash attention
// grid: (32 q-blocks of 128, 8 heads); 256 threads = 8 waves, 16 queries/wave.
// Per 64-key tile: S = Q*K^T (16 WMMA/wave), online softmax, O += P*V (16 WMMA/wave).
// K/V tiles double-buffered in LDS (90 KB total, WGP has 320 KB) and staged with
// async global->LDS loads one tile ahead.
#define KPAD 136  // K LDS row stride (halves)
#define VPAD 72   // V / P LDS row stride (halves)

__global__ __launch_bounds__(256) void mqa_attn(
    const _Float16* __restrict__ Q,   // [4096][1024]
    const _Float16* __restrict__ Kh,  // [4096][128]
    const _Float16* __restrict__ Vt,  // [128][4096]  (V transposed)
    float*          __restrict__ out) // [4096][1024]
{
    __shared__ _Float16 Kl[2][64 * KPAD];    // 2 x 17 KB : 64 keys x 128 d
    __shared__ _Float16 Vl[2][128 * VPAD];   // 2 x 18 KB : 128 d  x 64 keys
    __shared__ _Float16 Pl[8 * 16 * VPAD];   // 18 KB     : per-wave P 16x64

    const int tid  = threadIdx.x;
    const int lane = tid & 31;
    const int wave = tid >> 5;
    const int lr   = lane & 15;
    const int hi   = lane >> 4;
    const int h    = blockIdx.y;
    const int qw   = blockIdx.x * 128 + wave * 16;

    // Q fragments for this wave's 16 rows (A-fragment layout, loaded once)
    v16h qa[4];
    const _Float16* qrow = Q + (size_t)(qw + lr) * 1024 + h * 128;
#pragma unroll
    for (int kc = 0; kc < 4; kc++) {
        const _Float16* p = qrow + kc * 32 + (hi ? 8 : 0);
        qa[kc] = cat8(*(const v8h*)p, *(const v8h*)(p + 16));
    }

    v8f O[8];
#pragma unroll
    for (int d = 0; d < 8; d++) O[d] = (v8f){0.f,0.f,0.f,0.f,0.f,0.f,0.f,0.f};
    float Mr[8], Lr[8];
#pragma unroll
    for (int r = 0; r < 8; r++) { Mr[r] = -INFINITY; Lr[r] = 0.f; }
    const float scale = 0.08838834764831845f;  // 1/sqrt(128)

    // per-thread staging addresses
    const int krow = tid >> 2, kp = (tid & 3) * 32;
    const int vd   = tid >> 1, vp = (tid & 1) * 32;
    const _Float16* gk0 = Kh + (size_t)krow * 128 + kp;
    const _Float16* gv0 = Vt + (size_t)vd * 4096 + vp;

    auto stage = [&](int buf, int kt) {   // 4 async 16B loads per lane
        unsigned lk = lds_off(&Kl[buf][krow * KPAD + kp]);
        unsigned lv = lds_off(&Vl[buf][vd * VPAD + vp]);
        async_b128(lk,      gk0 + (size_t)kt * 128);
        async_b128(lk + 16, gk0 + (size_t)kt * 128 + 8);
        async_b128(lv,      gv0 + kt);
        async_b128(lv + 16, gv0 + kt + 8);
    };

    auto compute = [&](int buf) {
        // ---- S = scale * Q K^T : 4 n-subtiles x 4 k-chunks
        v8f S[4];
#pragma unroll
        for (int n = 0; n < 4; n++) {
            v8f s = (v8f){0.f,0.f,0.f,0.f,0.f,0.f,0.f,0.f};
            const _Float16* kb = &Kl[buf][(n * 16 + lr) * KPAD + (hi ? 8 : 0)];
#pragma unroll
            for (int kc = 0; kc < 4; kc++) {
                v16h b = cat8(*(const v8h*)(kb + kc * 32), *(const v8h*)(kb + kc * 32 + 16));
                s = __builtin_amdgcn_wmma_f32_16x16x32_f16(
                        false, qa[kc], false, b, (short)0, s, false, false);
            }
#pragma unroll
            for (int r = 0; r < 8; r++) s[r] *= scale;
            S[n] = s;
        }

        // ---- online softmax (row stats live in one VGPR per 16-lane half)
#pragma unroll
        for (int r = 0; r < 8; r++) {
            float m = fmaxf(fmaxf(S[0][r], S[1][r]), fmaxf(S[2][r], S[3][r]));
            m = fmaxf(m, __shfl_xor(m, 1, 32));
            m = fmaxf(m, __shfl_xor(m, 2, 32));
            m = fmaxf(m, __shfl_xor(m, 4, 32));
            m = fmaxf(m, __shfl_xor(m, 8, 32));
            const float newM  = fmaxf(Mr[r], m);
            const float alpha = __expf(Mr[r] - newM);
            Mr[r] = newM;
            float rs = 0.f;
#pragma unroll
            for (int n = 0; n < 4; n++) {
                float p = __expf(S[n][r] - newM);
                S[n][r] = p;
                rs += p;
            }
            rs += __shfl_xor(rs, 1, 32);
            rs += __shfl_xor(rs, 2, 32);
            rs += __shfl_xor(rs, 4, 32);
            rs += __shfl_xor(rs, 8, 32);
            Lr[r] = Lr[r] * alpha + rs;
#pragma unroll
            for (int d = 0; d < 8; d++) O[d][r] *= alpha;
        }

        // ---- P: C-layout -> wave-private LDS -> A-fragments
        _Float16* pw = Pl + wave * 16 * VPAD;
#pragma unroll
        for (int n = 0; n < 4; n++)
#pragma unroll
            for (int r = 0; r < 8; r++)
                pw[(hi * 8 + r) * VPAD + n * 16 + lr] = (_Float16)S[n][r];
        asm volatile("s_wait_dscnt 0x0" ::: "memory");

        v16h pa[2];
#pragma unroll
        for (int kc = 0; kc < 2; kc++) {
            const _Float16* p = pw + lr * VPAD + kc * 32 + (hi ? 8 : 0);
            pa[kc] = cat8(*(const v8h*)p, *(const v8h*)(p + 16));
        }

        // ---- O += P * V : 8 d-subtiles x 2 key-chunks
#pragma unroll
        for (int d = 0; d < 8; d++) {
            const _Float16* vb = &Vl[buf][(d * 16 + lr) * VPAD + (hi ? 8 : 0)];
#pragma unroll
            for (int kc = 0; kc < 2; kc++) {
                v16h b = cat8(*(const v8h*)(vb + kc * 32), *(const v8h*)(vb + kc * 32 + 16));
                O[d] = __builtin_amdgcn_wmma_f32_16x16x32_f16(
                           false, pa[kc], false, b, (short)0, O[d], false, false);
            }
        }
    };

    stage(0, 0);
    int buf = 0;
    for (int kt = 0; kt < 4096 - 64; kt += 64, buf ^= 1) {  // branch-free steady state
        stage(buf ^ 1, kt + 64);
        wait_async4();                    // current buffer's 4 loads done
        __syncthreads();
        compute(buf);
        __syncthreads();                  // reads done before buf is rewritten
    }
    wait_async0();                        // peeled last tile
    __syncthreads();
    compute(buf);

    // ---- epilogue: out = O / l  (fp32, coalesced across lanes)
#pragma unroll
    for (int r = 0; r < 8; r++) {
        const float inv = 1.0f / Lr[r];
        float* orow = out + (size_t)(qw + hi * 8 + r) * 1024 + h * 128;
#pragma unroll
        for (int d = 0; d < 8; d++)
            orow[d * 16 + lr] = O[d][r] * inv;
    }
}

// ---------------------------------------------------------------- launch
extern "C" void kernel_launch(void* const* d_in, const int* in_sizes, int n_in,
                              void* d_out, int out_size, void* d_ws, size_t ws_size,
                              hipStream_t stream) {
    const float* x  = (const float*)d_in[0];
    const float* Wq = (const float*)d_in[1];
    const float* bq = (const float*)d_in[2];
    const float* Wk = (const float*)d_in[3];
    const float* bk = (const float*)d_in[4];
    const float* Wv = (const float*)d_in[5];
    const float* bv = (const float*)d_in[6];
    float* out = (float*)d_out;

    // workspace layout (f16), ~21.5 MB total
    _Float16* xh  = (_Float16*)d_ws;
    _Float16* WqT = xh  + (size_t)4096 * 1024;
    _Float16* WkT = WqT + (size_t)1024 * 1024;
    _Float16* WvT = WkT + (size_t)128 * 1024;
    _Float16* Qh  = WvT + (size_t)128 * 1024;
    _Float16* Kh  = Qh  + (size_t)4096 * 1024;
    _Float16* Vt  = Kh  + (size_t)4096 * 128;

    convert_f16<<<16384, 256, 0, stream>>>(x, xh, 4096 * 1024);
    transpose_w<<<4096, 256, 0, stream>>>(Wq, WqT, 1024, 1024);
    transpose_w<<<512,  256, 0, stream>>>(Wk, WkT, 1024, 128);
    transpose_w<<<512,  256, 0, stream>>>(Wv, WvT, 1024, 128);

    gemm_qkv<<<dim3(8, 32), 256, 0, stream>>>(xh, WqT, bq, Qh, 4096, 1024, 1024, 0);
    gemm_qkv<<<dim3(1, 32), 256, 0, stream>>>(xh, WkT, bk, Kh, 4096, 128, 1024, 0);
    gemm_qkv<<<dim3(1, 32), 256, 0, stream>>>(xh, WvT, bv, Vt, 4096, 128, 1024, 1);

    mqa_attn<<<dim3(32, 8), 256, 0, stream>>>(Qh, Kh, Vt, out);
}
